// SwitchTransformersSparseMLP_3753801417119
// MI455X (gfx1250) — compile-verified, matrix operand order checked by reading, not run on hardware
//
#include <hip/hip_runtime.h>
#include <hip/hip_bf16.h>
#include <cmath>

// ---------------------------------------------------------------------------
// Switch-Transformer sparse MLP (top-1 MoE), MI455X / gfx1250, wave32.
//
// Roofline: expert GEMMs = 193 GFLOP over ~240MB unique HBM bytes (~10us
// @23.3TB/s) -> bf16 WMMA (v_wmma_f32_16x16x32_bf16, f32 accum). wi+wo in
// bf16 (75MB) are L2-resident (192MB). The binding resource is L2 fragment
// re-reads: M-tile = 32 rows/block so every 1KB B-fragment feeds 2 WMMAs,
// and the grid runs one expert at a time (e = slowest grid dim) so the live
// L2 set is one expert's 9.4MB. x-tiles are staged with the CDNA5 async DMA
// path (global_load_async_to_lds_b128 + s_wait_asynccnt).
// ---------------------------------------------------------------------------

typedef __attribute__((ext_vector_type(16))) __bf16 v16bf;
typedef __attribute__((ext_vector_type(8)))  float  v8f;
typedef unsigned short u16;
typedef unsigned int   u32;

#define D_MODEL 768
#define D_FF    3072
#define NEXP    8
#define CAP     320
#define NB      8
#define NS      2048
#define NTOK    (NB * NS)        // 16384

// d_out layout (floats): out [NTOK*768] | logits [NTOK*8] | expert_index [NTOK] (int bits)
#define OUT_ELEMS    (NTOK * D_MODEL)
#define LOGIT_ELEMS  (NTOK * NEXP)

// workspace byte offsets (all 256-aligned)
#define WS_HBF   0ull                               // h as bf16: NTOK*768*2      = 25,165,824
#define WS_WIP   25165824ull                        // wi packed bf16 fragments   = 37,748,736
#define WS_WOP   62914560ull                        // wo packed bf16 fragments   = 37,748,736
#define WS_SRC   100663296ull                       // src_idx int[NB*NEXP*CAP]   =     81,920
#define WS_MAXP  100745216ull                       // maxprob f32[NTOK]          =     65,536
#define WS_EIDX  100810752ull                       // argmax expert int[NTOK]    =     65,536
#define WS_KEPT  100876288ull                       // kept flag int[NTOK]        =     65,536

__device__ __forceinline__ u16 f32_to_bf16(float f) {
  union { float f; u32 u; } v; v.f = f;
  u32 r = v.u + 0x7FFFu + ((v.u >> 16) & 1u);   // round-to-nearest-even
  return (u16)(r >> 16);
}

__device__ __forceinline__ v16bf load_frag32B(const u16* p) {  // p 16B-aligned, 32B total
  v16bf r;
  ((uint4*)&r)[0] = ((const uint4*)p)[0];
  ((uint4*)&r)[1] = ((const uint4*)p)[1];
  return r;
}

// A-operand per documented 16-bit A 16x32 layout: lane m holds row m (m=lane&15),
// khalf=lane>>4; elements j=0..7 -> K = khalf*8+j, j=8..15 -> K = 16+khalf*8+(j-8).
__device__ __forceinline__ v16bf load_a_lds(const u16* row, int kk, int khalf) {
  v16bf a;
  ((uint4*)&a)[0] = *(const uint4*)(row + kk + khalf * 8);
  ((uint4*)&a)[1] = *(const uint4*)(row + kk + 16 + khalf * 8);
  return a;
}

// ---------------------------------------------------------------------------
// 1) Router: logits, softmax max-prob, argmax; also converts h rows to bf16.
// ---------------------------------------------------------------------------
__global__ __launch_bounds__(256) void router_kernel(
    const float* __restrict__ h, const float* __restrict__ wr,
    float* __restrict__ logits_out, u16* __restrict__ hbf,
    int* __restrict__ eidx, float* __restrict__ maxp) {
  const int token = blockIdx.x * 8 + (threadIdx.x >> 5);
  const int lane  = threadIdx.x & 31;
  const float* hr = h + (size_t)token * D_MODEL;
  u16* hbr = hbf + (size_t)token * D_MODEL;

  float acc[NEXP] = {0.f, 0.f, 0.f, 0.f, 0.f, 0.f, 0.f, 0.f};
  #pragma unroll 4
  for (int i = 0; i < D_MODEL / 32; ++i) {
    const int d = i * 32 + lane;
    const float hv = hr[d];
    hbr[d] = f32_to_bf16(hv);
    #pragma unroll
    for (int e = 0; e < NEXP; ++e) acc[e] += hv * wr[d * NEXP + e];
  }
  #pragma unroll
  for (int e = 0; e < NEXP; ++e) {
    #pragma unroll
    for (int off = 16; off > 0; off >>= 1) acc[e] += __shfl_xor(acc[e], off, 32);
  }
  float mx = acc[0]; int am = 0;
  #pragma unroll
  for (int e = 1; e < NEXP; ++e) if (acc[e] > mx) { mx = acc[e]; am = e; }
  float sum = 0.f;
  #pragma unroll
  for (int e = 0; e < NEXP; ++e) sum += __expf(acc[e] - mx);
  if (lane < NEXP) logits_out[(size_t)token * NEXP + lane] = acc[lane];
  if (lane == 0) { eidx[token] = am; maxp[token] = 1.0f / sum; }  // max prob = 1/sum
}

// ---------------------------------------------------------------------------
// 2) Pack a [E][K][N] f32 weight into bf16 WMMA B-fragments (lane-major 1KB).
// ---------------------------------------------------------------------------
__global__ __launch_bounds__(256) void pack_b_kernel(
    const float* __restrict__ src, u16* __restrict__ dst, int K, int N) {
  const int w = threadIdx.x >> 5, lane = threadIdx.x & 31;
  const int fid = blockIdx.x * 8 + w;
  const int ktiles = K >> 5;
  const int per_e  = (N >> 4) * ktiles;
  const int e = fid / per_e, rem = fid % per_e;
  const int ntile = rem / ktiles, ktile = rem % ktiles;
  const int n = ntile * 16 + (lane & 15);
  const int khalf = lane >> 4;
  const float* s = src + (size_t)e * K * N;
  u16* d = dst + (size_t)fid * 512 + lane * 16;
  #pragma unroll
  for (int j = 0; j < 16; ++j) {
    const int k = ktile * 32 + ((j < 8) ? (khalf * 8 + j) : (16 + khalf * 8 + (j - 8)));
    d[j] = f32_to_bf16(s[(size_t)k * N + n]);
  }
}

// ---------------------------------------------------------------------------
// 3) Dispatch: per-batch sequential capacity scan (matches cumsum semantics).
// ---------------------------------------------------------------------------
__global__ void dispatch_kernel(const int* __restrict__ eidx,
                                int* __restrict__ srcidx, int* __restrict__ kept,
                                int* __restrict__ eout) {
  const int b = blockIdx.x;
  for (int i = threadIdx.x; i < NEXP * CAP; i += blockDim.x)
    srcidx[b * NEXP * CAP + i] = -1;
  __syncthreads();
  if (threadIdx.x == 0) {
    int cnt[NEXP] = {0, 0, 0, 0, 0, 0, 0, 0};
    for (int s = 0; s < NS; ++s) {
      const int e = eidx[b * NS + s];
      const int p = cnt[e]++;
      const bool k = p < CAP;
      kept[b * NS + s] = k ? 1 : 0;
      eout[b * NS + s] = k ? e : 0;
      if (k) srcidx[(b * NEXP + e) * CAP + p] = s;
    }
  }
}

// ---------------------------------------------------------------------------
// 4) Fused expert FFN: y = relu(x @ wi[e]) @ wo[e], scaled scatter to out.
//    Block = 8 waves = one 32-row capacity tile (two 16-row halves / wave,
//    sharing every B-fragment -> halves L2 weight traffic vs M=16).
// ---------------------------------------------------------------------------
#define XSTR 776   // 768 + 8, multiple of 8 (keeps 16B alignment)
#define ASTR 136   // 128 + 8

__global__ __launch_bounds__(256) void moe_ffn_kernel(
    const u16* __restrict__ hbf, const u16* __restrict__ wip,
    const u16* __restrict__ wop, const int* __restrict__ srcidx,
    const float* __restrict__ maxp, float* __restrict__ out) {
  __shared__ __align__(16) u16 xs[32 * XSTR];   // x tile, bf16 (48KB + pad)
  __shared__ __align__(16) u16 as_[32 * ASTR];  // relu(x@wi) chunk, bf16
  __shared__ int   srcs[32];
  __shared__ float probs[32];

  const int ct = blockIdx.x, b = blockIdx.y, e = blockIdx.z;  // e slowest
  const int tid = threadIdx.x, w = tid >> 5, lane = tid & 31;
  const int khalf = lane >> 4, l16 = lane & 15;

  if (tid < 32) {
    const int s = srcidx[(b * NEXP + e) * CAP + ct * 32 + tid];
    srcs[tid]  = s;
    probs[tid] = (s >= 0) ? maxp[b * NS + s] : 0.f;
  }
  __syncthreads();

  { // stage x tile (32 rows x 768 bf16) via CDNA5 async DMA: 8 threads/row,
    // 12 x 16B async global->LDS transfers each; empty rows zero-filled in LDS.
    const int row = tid >> 3, c = tid & 7;
    const int s = srcs[row];
    u16* dstp = xs + row * XSTR + c * 96;
    if (s >= 0) {
      const u16* sp = hbf + ((size_t)(b * NS + s)) * D_MODEL + c * 96;
      #pragma unroll
      for (int i = 0; i < 12; ++i) {
        const u32 ldsoff = (u32)(uintptr_t)(dstp + i * 8);
        const u16* gp = sp + i * 8;
        asm volatile("global_load_async_to_lds_b128 %0, %1, off"
                     :: "v"(ldsoff), "v"(gp) : "memory");
      }
    } else {
      const uint4 z{0u, 0u, 0u, 0u};
      #pragma unroll
      for (int i = 0; i < 12; ++i) *(uint4*)(dstp + i * 8) = z;
    }
    asm volatile("s_wait_asynccnt 0x0" ::: "memory");
  }
  __syncthreads();

  v8f acc0[6], acc1[6];
  #pragma unroll
  for (int t = 0; t < 6; ++t) { acc0[t] = v8f{}; acc1[t] = v8f{}; }

  const u16* wi_e = wip + (size_t)e * (D_FF / 16) * (D_MODEL / 32) * 512;
  const u16* wo_e = wop + (size_t)e * (D_MODEL / 16) * (D_FF / 32) * 512;
  const u16* xrow0 = xs + l16 * XSTR;
  const u16* xrow1 = xs + (16 + l16) * XSTR;
  const u16* arow0 = as_ + l16 * ASTR;
  const u16* arow1 = as_ + (16 + l16) * ASTR;

  for (int f0 = 0; f0 < D_FF; f0 += 128) {
    // ---- stage 1: a = relu(x @ wi[:, f0+16w : f0+16w+16]), both row halves ----
    v8f aacc0 = v8f{}, aacc1 = v8f{};
    const u16* bptr = wi_e + ((size_t)((f0 >> 4) + w)) * (D_MODEL / 32) * 512 + lane * 16;
    #pragma unroll 4
    for (int kk = 0; kk < D_MODEL; kk += 32) {
      const v16bf bm = load_frag32B(bptr);
      __builtin_prefetch(bptr + 512, 0, 3);           // near-cache prefetch
      const v16bf a0 = load_a_lds(xrow0, kk, khalf);
      const v16bf a1 = load_a_lds(xrow1, kk, khalf);
      aacc0 = __builtin_amdgcn_wmma_f32_16x16x32_bf16(false, a0, false, bm,
                                                      (short)0, aacc0, false, false);
      aacc1 = __builtin_amdgcn_wmma_f32_16x16x32_bf16(false, a1, false, bm,
                                                      (short)0, aacc1, false, false);
      bptr += 512;
    }
    // C-layout -> LDS (relu + bf16)
    #pragma unroll
    for (int r = 0; r < 8; ++r) {
      float v0 = aacc0[r]; v0 = v0 > 0.f ? v0 : 0.f;
      float v1 = aacc1[r]; v1 = v1 > 0.f ? v1 : 0.f;
      as_[(khalf * 8 + r) * ASTR + w * 16 + l16]        = f32_to_bf16(v0);
      as_[(16 + khalf * 8 + r) * ASTR + w * 16 + l16]   = f32_to_bf16(v1);
    }
    __syncthreads();
    // ---- stage 2: acc += a(32x128) @ wo[f0:f0+128, w*96 : w*96+96] ----
    #pragma unroll
    for (int kk = 0; kk < 128; kk += 32) {
      const v16bf a20 = load_a_lds(arow0, kk, khalf);
      const v16bf a21 = load_a_lds(arow1, kk, khalf);
      const int ktile = (f0 + kk) >> 5;
      #pragma unroll
      for (int t = 0; t < 6; ++t) {
        const u16* bp = wo_e + ((size_t)((w * 6 + t) * (D_FF / 32) + ktile)) * 512 + lane * 16;
        const v16bf bm = load_frag32B(bp);
        acc0[t] = __builtin_amdgcn_wmma_f32_16x16x32_bf16(false, a20, false, bm,
                                                          (short)0, acc0[t], false, false);
        acc1[t] = __builtin_amdgcn_wmma_f32_16x16x32_bf16(false, a21, false, bm,
                                                          (short)0, acc1[t], false, false);
      }
    }
    __syncthreads();  // a_s reused next chunk
  }

  // epilogue: out[b, src, :] = maxprob * y  (kept rows only)
  #pragma unroll
  for (int t = 0; t < 6; ++t) {
    #pragma unroll
    for (int r = 0; r < 8; ++r) {
      const int m0 = khalf * 8 + r;
      const int m1 = 16 + m0;
      const int s0 = srcs[m0], s1 = srcs[m1];
      const int col = w * 96 + t * 16 + l16;
      if (s0 >= 0)
        out[((size_t)(b * NS + s0)) * D_MODEL + col] = probs[m0] * acc0[t][r];
      if (s1 >= 0)
        out[((size_t)(b * NS + s1)) * D_MODEL + col] = probs[m1] * acc1[t][r];
    }
  }
}

// ---------------------------------------------------------------------------
// 5) Dropped tokens: out = maxprob * h. One wave per token.
// ---------------------------------------------------------------------------
__global__ __launch_bounds__(256) void finalize_kernel(
    const float* __restrict__ h, const float* __restrict__ maxp,
    const int* __restrict__ kept, float* __restrict__ out) {
  const int token = blockIdx.x * 8 + (threadIdx.x >> 5);
  const int lane  = threadIdx.x & 31;
  if (kept[token]) return;  // wave-uniform
  const float p = maxp[token];
  const float* hr = h + (size_t)token * D_MODEL;
  float* orow = out + (size_t)token * D_MODEL;
  #pragma unroll 4
  for (int i = 0; i < D_MODEL / 32; ++i) orow[i * 32 + lane] = p * hr[i * 32 + lane];
}

// ---------------------------------------------------------------------------
extern "C" void kernel_launch(void* const* d_in, const int* in_sizes, int n_in,
                              void* d_out, int out_size, void* d_ws, size_t ws_size,
                              hipStream_t stream) {
  const float* h  = (const float*)d_in[0];
  const float* wr = (const float*)d_in[1];
  const float* wi = (const float*)d_in[2];
  const float* wo = (const float*)d_in[3];

  float* out_f      = (float*)d_out;
  float* logits_out = out_f + OUT_ELEMS;
  int*   eindex_out = (int*)(out_f + OUT_ELEMS + LOGIT_ELEMS);

  char* ws = (char*)d_ws;
  u16*   hbf  = (u16*)(ws + WS_HBF);
  u16*   wip  = (u16*)(ws + WS_WIP);
  u16*   wop  = (u16*)(ws + WS_WOP);
  int*   srcx = (int*)(ws + WS_SRC);
  float* maxp = (float*)(ws + WS_MAXP);
  int*   eidx = (int*)(ws + WS_EIDX);
  int*   kept = (int*)(ws + WS_KEPT);

  // 1) router (+ h -> bf16)
  router_kernel<<<NTOK / 8, 256, 0, stream>>>(h, wr, logits_out, hbf, eidx, maxp);
  // 2) weight repack into WMMA fragments (wi: K=768,N=3072; wo: K=3072,N=768)
  pack_b_kernel<<<(NEXP * (D_FF / 16) * (D_MODEL / 32)) / 8, 256, 0, stream>>>(wi, wip, D_MODEL, D_FF);
  pack_b_kernel<<<(NEXP * (D_MODEL / 16) * (D_FF / 32)) / 8, 256, 0, stream>>>(wo, wop, D_FF, D_MODEL);
  // 3) capacity scan
  dispatch_kernel<<<NB, 256, 0, stream>>>(eidx, srcx, kept, eindex_out);
  // 4) fused expert FFN (WMMA) with scaled scatter of kept tokens
  dim3 g(CAP / 32, NB, NEXP);
  moe_ffn_kernel<<<g, 256, 0, stream>>>(hbf, wip, wop, srcx, maxp, out_f);
  // 5) dropped tokens pass through
  finalize_kernel<<<NTOK / 8, 256, 0, stream>>>(h, maxp, kept, out_f);
}
